// CachedMultiHeadAttentionDecoderSelf_90245852824138
// MI455X (gfx1250) — matrix operand order repaired and checked by reading, not compiled, not run on hardware
//
#include <hip/hip_runtime.h>
#include <hip/hip_bf16.h>
#include <math.h>

// ---------------------------------------------------------------------------
// CachedMultiHeadAttentionDecoderSelf for MI455X (gfx1250, wave32, WMMA)
// B=8 T=512 S=2048 NS=1024 NH=16 HD=64, S+T=2560
// All matmuls via v_wmma_f32_16x16x32_f16; LDS staging via async-to-LDS
// copies (gfx1250 global_load_async_to_lds_b128 + s_wait_asynccnt).
// ---------------------------------------------------------------------------

#define Bb   8
#define Tt   512
#define Ss   2048
#define NS   1024
#define NH   16
#define HD   64
#define ST   2560            // S + T
#define MROWS (Bb * Tt)      // 4096
#define SCALE 0.35355339059327373f   // HD^(-0.25)

typedef _Float16 half_t;
typedef __attribute__((ext_vector_type(16))) _Float16 v16h;
typedef __attribute__((ext_vector_type(8)))  _Float16 v8h;
typedef __attribute__((ext_vector_type(8)))  float    v8f;
typedef int v4i_g __attribute__((vector_size(16)));   // matches builtin param

#define AS1 __attribute__((address_space(1)))
#define AS3 __attribute__((address_space(3)))

#if defined(__has_builtin)
#if __has_builtin(__builtin_amdgcn_global_load_async_to_lds_b128)
#define HAVE_ASYNC 1
#endif
#endif
#ifndef HAVE_ASYNC
#define HAVE_ASYNC 0
#endif

// 16-byte global -> LDS copy (async DMA path on gfx1250 when available).
__device__ __forceinline__ void cp16(half_t* lds_dst, const half_t* gsrc) {
#if HAVE_ASYNC
  __builtin_amdgcn_global_load_async_to_lds_b128(
      (AS1 v4i_g*)(uintptr_t)gsrc,
      (AS3 v4i_g*)(uint32_t)(uintptr_t)lds_dst, 0, 0);
#else
  *(v8h*)lds_dst = *(const v8h*)gsrc;
#endif
}

__device__ __forceinline__ void async_join() {
#if HAVE_ASYNC
#if defined(__has_builtin) && __has_builtin(__builtin_amdgcn_s_wait_asynccnt)
  __builtin_amdgcn_s_wait_asynccnt(0);
#else
  asm volatile("s_wait_asynccnt 0" ::: "memory");
#endif
#endif
}

// Assemble a 16-half WMMA fragment from two contiguous 16-byte chunks
// at p[0..7] and p[16..23] (CDNA5 A/B K-mapping when p is offset by grp*8
// within a 32-K row). p must be 16B aligned.
__device__ __forceinline__ v16h make_frag(const half_t* p) {
  v8h lo = *(const v8h*)(p);
  v8h hi = *(const v8h*)(p + 16);
  return __builtin_shufflevector(lo, hi, 0, 1, 2, 3, 4, 5, 6, 7,
                                         8, 9, 10, 11, 12, 13, 14, 15);
}

__device__ __forceinline__ v8f wmma_f16(v16h a, v16h b, v8f c) {
  return __builtin_amdgcn_wmma_f32_16x16x32_f16(false, a, false, b,
                                                (short)0, c, false, false);
}

// ---------------------------------------------------------------------------
// Kernel 0a: layer-0 KV cache -> f16 staging.
//   kh  : [B][2560][1024]  (K, pre-scaled; rows 0..2047 from cache)
//   vhT : [B][1024][2560]  (V transposed: PV B-fragments contiguous)
// ---------------------------------------------------------------------------
__global__ void cvt_cache_kernel(const float* __restrict__ kv,
                                 half_t* __restrict__ kh,
                                 half_t* __restrict__ vhT) {
  size_t idx = (size_t)blockIdx.x * blockDim.x + threadIdx.x;
  const size_t total = (size_t)Bb * Ss * NS;
  if (idx >= total) return;
  int n = (int)(idx & (NS - 1));
  size_t tmp = idx >> 10;
  int s = (int)(tmp & (Ss - 1));
  int b = (int)(tmp >> 11);
  size_t cbase = (size_t)b * 4 * 2 * Ss * NS;     // b stride = L*2*S*NS
  float kvl = kv[cbase + (size_t)s * NS + n];
  float vvl = kv[cbase + (size_t)Ss * NS + (size_t)s * NS + n];
  kh [((size_t)b * ST + s) * NS + n] = (half_t)(kvl * SCALE);
  vhT[((size_t)b * NS + n) * ST + s] = (half_t)vvl;
}

// Kernel 0b: x -> f16 row-major.
__global__ void cvt_x_kernel(const float* __restrict__ x,
                             half_t* __restrict__ xh) {
  size_t idx = (size_t)blockIdx.x * blockDim.x + threadIdx.x;
  if (idx >= (size_t)MROWS * NS) return;
  xh[idx] = (half_t)x[idx];
}

// Kernel 0c: weights -> f16 transposed WT[n][k].  z selects {Wq,Wk,Wv,Wo}.
__global__ void cvt_w_kernel(const float* __restrict__ Wq,
                             const float* __restrict__ Wk,
                             const float* __restrict__ Wv,
                             const float* __restrict__ Wo,
                             half_t* __restrict__ WT) {
  const int z = blockIdx.z;
  const float* W = (z == 0) ? Wq : (z == 1) ? Wk : (z == 2) ? Wv : Wo;
  size_t idx = (size_t)blockIdx.x * blockDim.x + threadIdx.x;
  if (idx >= (size_t)NS * NS) return;
  int n = (int)(idx & (NS - 1));
  int k = (int)(idx >> 10);
  WT[(size_t)z * NS * NS + (size_t)n * NS + k] = (half_t)W[idx];
}

// ---------------------------------------------------------------------------
// Unified WMMA GEMM:  C[4096,1024] = A(f16) @ W + bias, epilogue by mode.
// Block: 256 thr = 8 waves (2x4); block tile 64(M)x256(N); wave tile 32x64
// (2x4 subtiles -> 8 wmma per K-step); K-steps of 32; LDS-staged A/B tiles.
//   mode 0: q  -> qh  (f16, *SCALE)
//   mode 1: k  -> out_key (f32) + kh rows S..S+T (f16, *SCALE)
//   mode 2: v  -> out_val (f32) + vhT cols S..S+T (f16)
//   mode 3: o  -> out_attn (f32)
// ---------------------------------------------------------------------------
__global__ __launch_bounds__(256) void gemm_kernel(
    const half_t* __restrict__ A, const half_t* __restrict__ WTbase,
    const float* __restrict__ bq, const float* __restrict__ bv,
    const float* __restrict__ bo,
    float* __restrict__ out_key, float* __restrict__ out_val,
    float* __restrict__ out_attn,
    half_t* __restrict__ qh, half_t* __restrict__ kh,
    half_t* __restrict__ vhT, int modeBase) {
  const int mode = modeBase + blockIdx.z;
  const half_t* WT = WTbase + (size_t)mode * NS * NS;  // transposed [n][k]
  const int mBase = blockIdx.x * 64;
  const int nBase = blockIdx.y * 256;

  __shared__ __align__(16) half_t As[64][72];    // pitch 72 halves (144B)
  __shared__ __align__(16) half_t Bs[256][72];   // Bs[nlocal][k]

  const int tid = threadIdx.x;
  const int lane = tid & 31, wid = tid >> 5;
  const int wm = wid >> 2, wn = wid & 3;         // 2x4 wave grid
  const int grp = lane >> 4, hrow = lane & 15;

  v8f acc[2][4] = {{{}, {}, {}, {}}, {{}, {}, {}, {}}};

  for (int kb = 0; kb < NS; kb += 32) {
    {  // stage A tile: 64 rows x 32 halves = 256 x 16B chunks (1/thread)
      int r = tid >> 2, cc = (tid & 3) * 8;
      cp16(&As[r][cc], &A[(size_t)(mBase + r) * NS + kb + cc]);
    }
#pragma unroll
    for (int e = 0; e < 4; ++e) {  // stage B tile: 256 rows x 32 halves
      int c2 = tid + e * 256;
      int r = c2 >> 2, cc = (c2 & 3) * 8;
      cp16(&Bs[r][cc], &WT[(size_t)(nBase + r) * NS + kb + cc]);
    }
    async_join();
    __syncthreads();

    v16h a0 = make_frag(&As[wm * 32 + hrow][grp * 8]);
    v16h a1 = make_frag(&As[wm * 32 + 16 + hrow][grp * 8]);
#pragma unroll
    for (int t = 0; t < 4; ++t) {
      v16h b = make_frag(&Bs[wn * 64 + t * 16 + hrow][grp * 8]);
      acc[0][t] = wmma_f16(a0, b, acc[0][t]);
      acc[1][t] = wmma_f16(a1, b, acc[1][t]);
    }
    __syncthreads();
  }

#pragma unroll
  for (int s = 0; s < 2; ++s)
#pragma unroll
    for (int t = 0; t < 4; ++t) {
      const int col = nBase + wn * 64 + t * 16 + hrow;
      float bias_v = (mode == 0)   ? bq[col]
                     : (mode == 2) ? bv[col]
                     : (mode == 3) ? bo[col]
                                   : 0.0f;
#pragma unroll
      for (int i = 0; i < 8; ++i) {
        int row = mBase + wm * 32 + s * 16 + i + 8 * grp;
        float cv = acc[s][t][i] + bias_v;
        if (mode == 0) {
          qh[(size_t)row * NS + col] = (half_t)(cv * SCALE);
        } else if (mode == 1) {
          out_key[(size_t)row * NS + col] = cv;
          int b_ = row >> 9, tt = row & (Tt - 1);
          kh[((size_t)b_ * ST + Ss + tt) * NS + col] = (half_t)(cv * SCALE);
        } else if (mode == 2) {
          out_val[(size_t)row * NS + col] = cv;
          int b_ = row >> 9, tt = row & (Tt - 1);
          vhT[((size_t)b_ * NS + col) * ST + Ss + tt] = (half_t)cv;
        } else {
          out_attn[(size_t)row * NS + col] = cv;
        }
      }
    }
}

// ---------------------------------------------------------------------------
// Attention.  grid = (T/64, NH, B), block = 128 thr = 4 waves, all waves
// share (b,h) so K/V tiles are staged to LDS once per block.  Each wave owns
// 16 query rows: two-pass streaming softmax with the reference's
// qk==0 -> -inf masking, P.V accumulated via WMMA.
// ---------------------------------------------------------------------------
__global__ __launch_bounds__(128) void attn_kernel(
    const half_t* __restrict__ qh, const half_t* __restrict__ kh,
    const half_t* __restrict__ vhT, half_t* __restrict__ wvh) {
  const int b = blockIdx.z, h = blockIdx.y;
  const int tid = threadIdx.x, lane = tid & 31, wid = tid >> 5;
  const int t0 = blockIdx.x * 64 + wid * 16;
  const int grp = lane >> 4, hrow = lane & 15;

  __shared__ __align__(16) half_t Ks[32][72];    // 32 keys x 64 d
  __shared__ __align__(16) half_t Vs[64][40];    // 64 d x 32 keys
  __shared__ __align__(16) half_t Pt[4][16][40]; // per-wave P tiles (16x32)

  const half_t* pq = &qh[((size_t)(b * Tt + t0 + hrow)) * NS + h * HD];
  v16h aq0 = make_frag(pq + grp * 8);        // q rows, d 0..31 (pre-scaled)
  v16h aq1 = make_frag(pq + 32 + grp * 8);   // d 32..63

  const half_t* kbase = &kh[(size_t)b * ST * NS + h * HD];
  const half_t* vbase = &vhT[((size_t)b * NS + h * HD) * ST];

  // ---- pass A: per-row max over all 2560 keys ----
  float rowMax[8];
#pragma unroll
  for (int i = 0; i < 8; ++i) rowMax[i] = -INFINITY;
  for (int j2 = 0; j2 < ST / 32; ++j2) {
#pragma unroll
    for (int e = 0; e < 2; ++e) {  // stage K tile: 32 rows x 8 chunks
      int c2 = tid + e * 128;
      int r = c2 >> 3, cc = (c2 & 7) * 8;
      cp16(&Ks[r][cc], kbase + (size_t)(j2 * 32 + r) * NS + cc);
    }
    async_join();
    __syncthreads();
#pragma unroll
    for (int hb = 0; hb < 2; ++hb) {
      v8f c = {};
      c = wmma_f16(aq0, make_frag(&Ks[hb * 16 + hrow][grp * 8]), c);
      c = wmma_f16(aq1, make_frag(&Ks[hb * 16 + hrow][32 + grp * 8]), c);
#pragma unroll
      for (int i = 0; i < 8; ++i) {
        float s = c[i];
        if (s != 0.0f) rowMax[i] = fmaxf(rowMax[i], s);  // mask zeros
      }
    }
    __syncthreads();
  }
#pragma unroll
  for (int i = 0; i < 8; ++i)
#pragma unroll
    for (int m = 1; m < 16; m <<= 1)
      rowMax[i] = fmaxf(rowMax[i], __shfl_xor(rowMax[i], m, 32));

  // ---- pass B: recompute scores, exp, accumulate sum and P.V ----
  float rowSum[8];
#pragma unroll
  for (int i = 0; i < 8; ++i) rowSum[i] = 0.0f;
  v8f o[4] = {{}, {}, {}, {}};

  for (int j2 = 0; j2 < ST / 32; ++j2) {
#pragma unroll
    for (int e = 0; e < 2; ++e) {  // K tile
      int c2 = tid + e * 128;
      int r = c2 >> 3, cc = (c2 & 7) * 8;
      cp16(&Ks[r][cc], kbase + (size_t)(j2 * 32 + r) * NS + cc);
    }
#pragma unroll
    for (int e = 0; e < 2; ++e) {  // V tile: 64 rows x 4 chunks
      int c2 = tid + e * 128;
      int r = c2 >> 2, cc = (c2 & 3) * 8;
      cp16(&Vs[r][cc], vbase + (size_t)r * ST + j2 * 32 + cc);
    }
    async_join();
    __syncthreads();
#pragma unroll
    for (int hb = 0; hb < 2; ++hb) {
      v8f c = {};
      c = wmma_f16(aq0, make_frag(&Ks[hb * 16 + hrow][grp * 8]), c);
      c = wmma_f16(aq1, make_frag(&Ks[hb * 16 + hrow][32 + grp * 8]), c);
#pragma unroll
      for (int i = 0; i < 8; ++i) {
        float s = c[i];
        float p = (s == 0.0f) ? 0.0f : __expf(s - rowMax[i]);
        rowSum[i] += p;
        Pt[wid][i + 8 * grp][hb * 16 + hrow] = (half_t)p;
      }
    }
    v16h ap = make_frag(&Pt[wid][hrow][grp * 8]);  // C-layout -> A-layout
#pragma unroll
    for (int d = 0; d < 4; ++d) {
      v16h bvf = make_frag(&Vs[d * 16 + hrow][grp * 8]);
      o[d] = wmma_f16(ap, bvf, o[d]);
    }
    __syncthreads();
  }

#pragma unroll
  for (int i = 0; i < 8; ++i)
#pragma unroll
    for (int m = 1; m < 16; m <<= 1)
      rowSum[i] += __shfl_xor(rowSum[i], m, 32);

#pragma unroll
  for (int d = 0; d < 4; ++d)
#pragma unroll
    for (int i = 0; i < 8; ++i) {
      int trow = t0 + i + 8 * grp;
      float ov = o[d][i] / rowSum[i];
      wvh[((size_t)(b * Tt + trow)) * NS + h * HD + d * 16 + hrow] =
          (half_t)ov;
    }
}

// ---------------------------------------------------------------------------
extern "C" void kernel_launch(void* const* d_in, const int* in_sizes, int n_in,
                              void* d_out, int out_size, void* d_ws,
                              size_t ws_size, hipStream_t stream) {
  const float* x  = (const float*)d_in[0];
  const float* kv = (const float*)d_in[1];
  // d_in[2] = offset (unused by the reference forward)
  const float* Wq = (const float*)d_in[3];
  const float* bq = (const float*)d_in[4];
  const float* Wk = (const float*)d_in[5];
  const float* Wv = (const float*)d_in[6];
  const float* bv = (const float*)d_in[7];
  const float* Wo = (const float*)d_in[8];
  const float* bo = (const float*)d_in[9];

  float* out_attn = (float*)d_out;                  // [B,T,NS]
  float* out_key  = out_attn + (size_t)MROWS * NS;  // [B,T,NS]
  float* out_val  = out_key + (size_t)MROWS * NS;   // [B,T,NS]

  // Workspace (f16): xh | WT(4 mats) | qh | kh | vhT | wvh  (~120 MiB)
  half_t* xh  = (half_t*)d_ws;
  half_t* WT  = xh + (size_t)MROWS * NS;
  half_t* qh  = WT + (size_t)4 * NS * NS;
  half_t* kh  = qh + (size_t)MROWS * NS;            // [B][2560][1024]
  half_t* vhT = kh + (size_t)Bb * ST * NS;          // [B][1024][2560]
  half_t* wvh = vhT + (size_t)Bb * ST * NS;         // [B,T,NS]

  {
    size_t total = (size_t)Bb * Ss * NS;
    cvt_cache_kernel<<<(int)((total + 255) / 256), 256, 0, stream>>>(kv, kh,
                                                                     vhT);
  }
  {
    size_t total = (size_t)MROWS * NS;
    cvt_x_kernel<<<(int)((total + 255) / 256), 256, 0, stream>>>(x, xh);
  }
  {
    dim3 grid((NS * NS + 255) / 256, 1, 4);
    cvt_w_kernel<<<grid, 256, 0, stream>>>(Wq, Wk, Wv, Wo, WT);
  }
  {
    dim3 grid(MROWS / 64, NS / 256, 3);   // QKV projections
    gemm_kernel<<<grid, 256, 0, stream>>>(xh, WT, bq, bv, bo, out_key,
                                          out_val, out_attn, qh, kh, vhT, 0);
  }
  {
    dim3 grid(Tt / 64, NH, Bb);
    attn_kernel<<<grid, 128, 0, stream>>>(qh, kh, vhT, wvh);
  }
  {
    dim3 grid(MROWS / 64, NS / 256, 1);   // output projection
    gemm_kernel<<<grid, 256, 0, stream>>>(wvh, WT, bq, bv, bo, out_key,
                                          out_val, out_attn, qh, kh, vhT, 3);
  }
}
